// AttnBlock_56083682951596
// MI455X (gfx1250) — matrix operand order repaired and checked by reading, not compiled, not run on hardware
//
#include <hip/hip_runtime.h>
#include <hip/hip_fp16.h>

typedef _Float16 f16;
typedef _Float16 v16h __attribute__((ext_vector_type(16)));
typedef _Float16 v8h  __attribute__((ext_vector_type(8)));
typedef float    v8f  __attribute__((ext_vector_type(8)));

#define DIMC 512
#define HW   4096   // 64*64 spatial positions
#define NP   64
#define NQ   64
#define ATTN_SCALE 0.125f

// ---------------------------------------------------------------------------
// K1: column sums of Wq/Wk (for collapsed q/k projections) + bias sums
// ---------------------------------------------------------------------------
__global__ void colsum_bias_kernel(const float* __restrict__ Wq, const float* __restrict__ bq,
                                   const float* __restrict__ Wk, const float* __restrict__ bk,
                                   float* __restrict__ wqs, float* __restrict__ wks,
                                   float* __restrict__ bs) {
  const int t = threadIdx.x;
  const float* W = (blockIdx.x == 0) ? Wq : Wk;
  const float* b = (blockIdx.x == 0) ? bq : bk;
  float* o       = (blockIdx.x == 0) ? wqs : wks;
  float s = 0.f;
  for (int r = 0; r < DIMC; ++r) s += W[r * DIMC + t];
  o[t] = s;
  __shared__ float sb[DIMC];
  sb[t] = b[t];
  __syncthreads();
  for (int off = DIMC / 2; off > 0; off >>= 1) {
    if (t < off) sb[t] += sb[t + off];
    __syncthreads();
  }
  if (t == 0) bs[blockIdx.x] = sb[0];
}

// ---------------------------------------------------------------------------
// K2: f32 -> f16 conversion (x and Wv)
// ---------------------------------------------------------------------------
__global__ void cvt_kernel(const float* __restrict__ src, f16* __restrict__ dst, int n) {
  int i = blockIdx.x * blockDim.x + threadIdx.x;
  if (i < n) dst[i] = (f16)src[i];
}

// ---------------------------------------------------------------------------
// K3: qs[hw] = x[hw,:].wqs + bqsum ; ks[hw] = x[hw,:].wks + bksum
// one wave per hw position, wave32 shuffle reduction
// ---------------------------------------------------------------------------
__global__ void qsks_kernel(const float* __restrict__ x, const float* __restrict__ wqs,
                            const float* __restrict__ wks, const float* __restrict__ bs,
                            float* __restrict__ qs, float* __restrict__ ks) {
  const int lane = threadIdx.x & 31;
  const int wave = threadIdx.x >> 5;
  const int hw = blockIdx.x * 8 + wave;
  const float* xr = x + (size_t)hw * DIMC;
  float sq = 0.f, sk = 0.f;
  for (int i = lane; i < DIMC; i += 32) {
    float xv = xr[i];
    sq = fmaf(xv, wqs[i], sq);
    sk = fmaf(xv, wks[i], sk);
  }
  for (int off = 16; off > 0; off >>= 1) {
    sq += __shfl_down(sq, off);
    sk += __shfl_down(sk, off);
  }
  if (lane == 0) {
    qs[hw] = sq + bs[0];
    ks[hw] = sk + bs[1];
  }
}

// ---------------------------------------------------------------------------
// K4: per-p min/max of ks row (analytic softmax max)
// ---------------------------------------------------------------------------
__global__ void kminmax_kernel(const float* __restrict__ ks, float* __restrict__ kmax,
                               float* __restrict__ kmin) {
  int p = threadIdx.x;
  float mx = -3.402823466e38f, mn = 3.402823466e38f;
  for (int q = 0; q < NQ; ++q) {
    float v = ks[p * NQ + q];
    mx = fmaxf(mx, v);
    mn = fminf(mn, v);
  }
  kmax[p] = mx;
  kmin[p] = mn;
}

// ---------------------------------------------------------------------------
// K5: V projection via WMMA f16: V[hw,d] = sum_c x[hw,c]*Wv[d,c] + bv[d]
// stored TRANSPOSED as f16: vt[d*4096 + hw]  (makes B contiguous in K7)
// one wave = one 16x16 output tile, K-loop over 512 channels in steps of 32
// ---------------------------------------------------------------------------
__global__ void __launch_bounds__(256) vproj_kernel(const f16* __restrict__ xh,
                                                    const f16* __restrict__ wvh,
                                                    const float* __restrict__ bv,
                                                    f16* __restrict__ vt) {
  const int lane = threadIdx.x & 31;
  const int wave = threadIdx.x >> 5;
  const int gw = blockIdx.x * 8 + wave;
  const int hwt = gw >> 5, dt = gw & 31;
  const int hw0 = hwt * 16, d0 = dt * 16;
  const int n = lane & 15;
  const int hi = lane >> 4;
  // A (16-bit 16x32): lane row M=n; K set = {hi*8 .. +7} u {16+hi*8 .. +7}
  const f16* arow = xh + (size_t)(hw0 + n) * DIMC + hi * 8;
  // B (16-bit 32x16): lane col N=n; K set = {hi*8 .. +7} u {16+hi*8 .. +7}
  const f16* brow = wvh + (size_t)(d0 + n) * DIMC + hi * 8;
  float bias = bv[d0 + n];
  v8f acc = {bias, bias, bias, bias, bias, bias, bias, bias};
#pragma unroll
  for (int c0 = 0; c0 < DIMC; c0 += 32) {
    v8h alo = *(const v8h*)(arow + c0);
    v8h ahi = *(const v8h*)(arow + c0 + 16);
    v8h blo = *(const v8h*)(brow + c0);
    v8h bhi = *(const v8h*)(brow + c0 + 16);
    v16h A = __builtin_shufflevector(alo, ahi, 0, 1, 2, 3, 4, 5, 6, 7, 8, 9, 10, 11, 12, 13, 14, 15);
    v16h B = __builtin_shufflevector(blo, bhi, 0, 1, 2, 3, 4, 5, 6, 7, 8, 9, 10, 11, 12, 13, 14, 15);
    acc = __builtin_amdgcn_wmma_f32_16x16x32_f16(false, A, false, B, (short)0, acc, false, false);
  }
#pragma unroll
  for (int r = 0; r < 8; ++r) {
    // C/D layout: VGPR r -> row M = r + hi*8, col N = n
    vt[(size_t)(d0 + n) * HW + (hw0 + r + hi * 8)] = (f16)acc[r];
  }
}

// ---------------------------------------------------------------------------
// K6: softmax max + inverse denominator per (p, hw)
// m = a>=0 ? a*max(ks[p]) : a*min(ks[p]) ; inv = 1/sum_q exp(a*ks[p,q]-m)
// layout [p][hw] so K7 reads coalesced
// ---------------------------------------------------------------------------
__global__ void mden_kernel(const float* __restrict__ qs, const float* __restrict__ ks,
                            const float* __restrict__ kmax, const float* __restrict__ kmin,
                            float* __restrict__ m_out, float* __restrict__ inv_out) {
  int idx = blockIdx.x * blockDim.x + threadIdx.x;
  int p = idx >> 12;           // 4096 hw per p -> p uniform per block (scalar ks loads)
  int hw = idx & (HW - 1);
  float a = ATTN_SCALE * qs[hw];
  float m = (a >= 0.f) ? a * kmax[p] : a * kmin[p];
  const float* kr = ks + p * NQ;
  float s = 0.f;
#pragma unroll 8
  for (int q = 0; q < NQ; ++q) s += __expf(fmaf(a, kr[q], -m));
  m_out[p * HW + hw] = m;
  inv_out[p * HW + hw] = 1.f / s;
}

// ---------------------------------------------------------------------------
// K7: main attention GEMM. Per block: one p, 32 hw rows, all 512 d.
// out[hw,p,d] = sum_q softmax_q(a_hw*ks[p,q]) * v[p,q,d]
// A built in-register from v_exp_f32 (reused over 4 d-tiles), B = vt (f16,
// contiguous per-lane columns, 2x b128 per 32-K step), NT f32 stores.
// ---------------------------------------------------------------------------
__global__ void __launch_bounds__(256) attn_kernel(const float* __restrict__ qs,
                                                   const float* __restrict__ ks,
                                                   const float* __restrict__ m_in,
                                                   const float* __restrict__ inv_in,
                                                   const f16* __restrict__ vt,
                                                   float* __restrict__ out) {
  const int p = blockIdx.x & 63;
  const int hwg = blockIdx.x >> 6;
  const int hw0 = hwg * 32;
  __shared__ float s_ks[NQ];
  __shared__ float s_a[32], s_m[32], s_i[32];
  const int tid = threadIdx.x;
  if (tid < NQ) s_ks[tid] = ks[p * NQ + tid];
  if (tid < 32) {
    int row = hw0 + tid;
    s_a[tid] = ATTN_SCALE * qs[row];
    s_m[tid] = m_in[p * HW + row];
    s_i[tid] = inv_in[p * HW + row];
  }
  __syncthreads();
  const int lane = tid & 31;
  const int wave = tid >> 5;
  const int n = lane & 15;
  const int hi = lane >> 4;
  const int dbase = wave * 64;  // each wave owns 64 d columns (4 tiles of 16)
  const float a0 = s_a[n],      m0 = s_m[n],      i0 = s_i[n];
  const float a1 = s_a[16 + n], m1 = s_m[16 + n], i1 = s_i[16 + n];
  v8f C[2][4] = {};
#pragma unroll
  for (int s = 0; s < 2; ++s) {
    const int q0 = s * 32;
    const int ab = hi * 16;
    float kv[16];
#pragma unroll
    for (int j = 0; j < 8; ++j) {
      kv[j]     = s_ks[q0 + ab + j];
      kv[8 + j] = s_ks[q0 + ab + 8 + j];
    }
    v16h A0{}, A1{};
#pragma unroll
    for (int j = 0; j < 16; ++j) {
      A0[j] = (f16)(__expf(fmaf(a0, kv[j], -m0)) * i0);
      A1[j] = (f16)(__expf(fmaf(a1, kv[j], -m1)) * i1);
    }
#pragma unroll
    for (int dt = 0; dt < 4; ++dt) {
      const f16* bp = vt + (size_t)(dbase + dt * 16 + n) * HW + p * NQ + q0 + hi * 16;
      v8h blo = *(const v8h*)bp;
      v8h bhi = *(const v8h*)(bp + 8);
      v16h B = __builtin_shufflevector(blo, bhi, 0, 1, 2, 3, 4, 5, 6, 7, 8, 9, 10, 11, 12, 13, 14, 15);
      C[0][dt] = __builtin_amdgcn_wmma_f32_16x16x32_f16(false, A0, false, B, (short)0, C[0][dt], false, false);
      C[1][dt] = __builtin_amdgcn_wmma_f32_16x16x32_f16(false, A1, false, B, (short)0, C[1][dt], false, false);
    }
  }
#pragma unroll
  for (int rb = 0; rb < 2; ++rb) {
#pragma unroll
    for (int dt = 0; dt < 4; ++dt) {
#pragma unroll
      for (int r = 0; r < 8; ++r) {
        int hwrow = hw0 + rb * 16 + r + hi * 8;
        int d = dbase + dt * 16 + n;
        __builtin_nontemporal_store(C[rb][dt][r], out + ((size_t)hwrow * NP + p) * DIMC + d);
      }
    }
  }
}

// ---------------------------------------------------------------------------
// host launcher
// ---------------------------------------------------------------------------
extern "C" void kernel_launch(void* const* d_in, const int* in_sizes, int n_in,
                              void* d_out, int out_size, void* d_ws, size_t ws_size,
                              hipStream_t stream) {
  const float* x  = (const float*)d_in[0];
  // d_in[1] = mask sentinel (unused, mask is None)
  const float* Wq = (const float*)d_in[2];
  const float* bq = (const float*)d_in[3];
  const float* Wk = (const float*)d_in[4];
  const float* bk = (const float*)d_in[5];
  const float* Wv = (const float*)d_in[6];
  const float* bv = (const float*)d_in[7];
  float* out = (float*)d_out;

  // workspace layout
  float* WQS  = (float*)d_ws;          // 512
  float* WKS  = WQS + 512;             // 512
  float* BS   = WKS + 512;             // 16 (2 used: bqsum, bksum)
  float* QS   = BS + 16;               // 4096
  float* KS   = QS + 4096;             // 4096
  float* KMAX = KS + 4096;             // 64
  float* KMIN = KMAX + 64;             // 64
  float* Mst  = KMIN + 64;             // 262144  [p][hw]
  float* INV  = Mst + 262144;          // 262144  [p][hw]
  f16* XH  = (f16*)(INV + 262144);     // 2097152 [hw][c]
  f16* WVH = XH + 2097152;             // 262144  [d][c]
  f16* VT  = WVH + 262144;             // 2097152 [d][pq]

  colsum_bias_kernel<<<2, 512, 0, stream>>>(Wq, bq, Wk, bk, WQS, WKS, BS);
  cvt_kernel<<<8192, 256, 0, stream>>>(x, XH, 2097152);
  cvt_kernel<<<1024, 256, 0, stream>>>(Wv, WVH, 262144);
  qsks_kernel<<<512, 256, 0, stream>>>(x, WQS, WKS, BS, QS, KS);
  kminmax_kernel<<<1, 64, 0, stream>>>(KS, KMAX, KMIN);
  vproj_kernel<<<1024, 256, 0, stream>>>(XH, WVH, bv, VT);
  mden_kernel<<<1024, 256, 0, stream>>>(QS, KS, KMAX, KMIN, Mst, INV);
  attn_kernel<<<8192, 256, 0, stream>>>(QS, KS, Mst, INV, VT, out);
}
